// scene_graph_with_monet_v1_62878321213569
// MI455X (gfx1250) — matrix operand order repaired and checked by reading, not compiled, not run on hardware
//
#include <hip/hip_runtime.h>

typedef __attribute__((ext_vector_type(16))) __bf16 v16bf;
typedef __attribute__((ext_vector_type(8)))  __bf16 v8bf;
typedef __attribute__((ext_vector_type(8)))  float  v8f;

#define BN   4
#define SN   11
#define PIX  384      // 16*24
#define FD   256
#define CRM  640      // 256 ctx + 384 rel rows
#define KOBJ 512
#define KREL 896
#define KFC  98304    // 256*384
#define OD   256

// ---------------------------------------------------------------------------
// 1) Antialiased linear resample along one axis (jax.image.resize bilinear,
//    antialias=True downsampling: triangle kernel, support = in/out,
//    weights renormalized over in-range taps).
// ---------------------------------------------------------------------------
__global__ void resample_kernel(const float* __restrict__ src, float* __restrict__ dst,
                                int outer, int I, int O, int inner) {
    int idx = blockIdx.x * blockDim.x + threadIdx.x;
    int total = outer * O * inner;
    if (idx >= total) return;
    int in_i = idx % inner;
    int o    = (idx / inner) % O;
    int ou   = idx / (inner * O);
    float scale = (float)I / (float)O;             // > 1 (downsample)
    float c = (o + 0.5f) * scale - 0.5f;
    int lo = (int)ceilf(c - scale);
    int hi = (int)floorf(c + scale);
    if (lo < 0) lo = 0;
    if (hi > I - 1) hi = I - 1;
    float acc = 0.f, wsum = 0.f;
    for (int i = lo; i <= hi; ++i) {
        float w = 1.0f - fabsf((float)i - c) / scale;
        if (w < 0.f) w = 0.f;
        acc  += w * src[((size_t)ou * I + i) * inner + in_i];
        wsum += w;
    }
    dst[idx] = acc / wsum;
}

// ---------------------------------------------------------------------------
// 2) f32 -> bf16 weight packing
// ---------------------------------------------------------------------------
__global__ void pack_bf16_kernel(const float* __restrict__ s, __bf16* __restrict__ d, int n) {
    int i = blockIdx.x * blockDim.x + threadIdx.x;
    if (i < n) d[i] = (__bf16)s[i];
}

__device__ __forceinline__ v16bf cat8(v8bf lo, v8bf hi) {
    return __builtin_shufflevector(lo, hi, 0,1,2,3,4,5,6,7,8,9,10,11,12,13,14,15);
}
__device__ __forceinline__ float4 mul4(float4 a, float4 b) {
    return make_float4(a.x*b.x, a.y*b.y, a.z*b.z, a.w*b.w);
}
__device__ __forceinline__ float4 add4(float4 a, float4 b) {
    return make_float4(a.x+b.x, a.y+b.y, a.z+b.z, a.w+b.w);
}

// ---------------------------------------------------------------------------
// 3) Templated WMMA GEMM.  128 threads = 4 waves; each wave owns MT 16x16
//    C tiles (MT M-tiles); WG stages a 64k x 16n B tile per sync round, stored
//    [n][k] bf16 so B-fragments are aligned b128 LDS loads.  2*MT WMMAs per
//    barrier pair; B-fragment reused across all MT M-tiles of the wave.
//
//    MODE 0 (MT=2): ctx+rel conv  A=Wcr(640x256)    B=inp[z]                          -> ctxrel f32
//    MODE 1 (MT=4): obj conv      A=Wobjf(256x512)  B=[inp*m;ctx*m;ctx*m^2] (z=b*S+s) -> comb_obj f32 + relu bf16 flat
//    MODE 2 (MT=4): rel conv      A=Wrelf(256x896)  B=[cobj_s;cobj_o;rel*pm;..] (z=pair) -> relu bf16 flat
//    MODE 3 (MT=4): FC split-K    A=Wfc(256x98304)  B=flat k-major (z=k-chunk)        -> atomicAdd f32 acc
// ---------------------------------------------------------------------------
template <int MODE, int MT>
__global__ void gemm_wmma_kernel(
    int batch,
    const __bf16* __restrict__ A, int K,
    const float* __restrict__ bias, const float* __restrict__ bias2,
    const float* __restrict__ inp,
    const float* __restrict__ ctxrel,
    const float* __restrict__ comb_obj,
    const float* __restrict__ mres,
    const __bf16* __restrict__ flat,
    float* __restrict__ outf,
    __bf16* __restrict__ outh,
    int kcsz, int nvalid)
{
    __shared__ __bf16 bt[16][72];   // [n][k0..63], stride 144B (16B aligned, conflict-free)

    const int lane = threadIdx.x & 31;
    const int wave = threadIdx.x >> 5;
    const int n0   = blockIdx.x * 16;
    const int m0   = (blockIdx.y * 4 * MT + wave * MT) * 16;  // wave covers MT*16 rows
    const int z    = blockIdx.z;
    const int bb   = (MODE == 0) ? z : batch;

    int ks, ke;
    if (MODE <= 2) { ks = 0; ke = K; }
    else           { ks = z * kcsz; ke = ks + kcsz; }

    v8f acc[MT];
#pragma unroll
    for (int t = 0; t < MT; ++t) acc[t] = (v8f){};

    for (int kk = ks; kk < ke; kk += 64) {
        __syncthreads();
        // ----------------- stage B tile (64 x 16) into LDS [n][k] -----------------
        if (MODE <= 2) {
            int kq = threadIdx.x >> 2;          // 0..31
            int ni = (threadIdx.x & 3) << 2;    // 0,4,8,12
            int p  = n0 + ni;
#pragma unroll
            for (int kh = 0; kh < 2; ++kh) {
                int k = kh * 32 + kq;
                int c = kk + k;
                float4 v;
                if (MODE == 0) {
                    v = *(const float4*)(inp + ((size_t)(bb * FD + c)) * PIX + p);
                } else if (MODE == 1) {
                    int bz = z / SN;
                    float4 m4 = *(const float4*)(mres + (size_t)z * PIX + p);
                    if (c < 256) {
                        v = *(const float4*)(inp + ((size_t)(bz * FD + c)) * PIX + p);
                        v = mul4(v, m4);
                    } else {
                        v = *(const float4*)(ctxrel + ((size_t)(bz * CRM + (c - 256))) * PIX + p);
                        v = mul4(v, m4);
                        if (c >= 384) v = mul4(v, m4);
                    }
                } else { // MODE 2
                    int sub = z / SN, obj = z % SN;
                    float4 ms = *(const float4*)(mres + ((size_t)(bb * SN + sub)) * PIX + p);
                    float4 mo = *(const float4*)(mres + ((size_t)(bb * SN + obj)) * PIX + p);
                    if (c < 256) {
                        v = *(const float4*)(comb_obj + ((size_t)((bb * SN + sub) * OD + c)) * PIX + p);
                    } else if (c < 512) {
                        v = *(const float4*)(comb_obj + ((size_t)((bb * SN + obj) * OD + (c - 256))) * PIX + p);
                    } else {
                        float4 pm = add4(ms, mo);
                        if (c < 640) {
                            v = *(const float4*)(ctxrel + ((size_t)(bb * CRM + 256 + (c - 512))) * PIX + p);
                            v = mul4(v, pm);
                        } else if (c < 768) {
                            v = *(const float4*)(ctxrel + ((size_t)(bb * CRM + 384 + (c - 640))) * PIX + p);
                            v = mul4(mul4(v, pm), ms);
                        } else {
                            v = *(const float4*)(ctxrel + ((size_t)(bb * CRM + 512 + (c - 768))) * PIX + p);
                            v = mul4(mul4(v, pm), mo);
                        }
                    }
                }
                bt[ni + 0][k] = (__bf16)v.x;
                bt[ni + 1][k] = (__bf16)v.y;
                bt[ni + 2][k] = (__bf16)v.z;
                bt[ni + 3][k] = (__bf16)v.w;
            }
        } else {
            // FC: B already bf16 k-major: flat[n][k]; 8 threads per row, 16B each
            int n  = threadIdx.x >> 3;          // 0..15
            int j  = (threadIdx.x & 7) << 3;    // 0,8,..,56
            int ng = n0 + n;
            v8bf val;
            if (ng < nvalid) {
                val = *(const v8bf*)(flat + (size_t)ng * K + kk + j);
            } else {
#pragma unroll
                for (int q = 0; q < 8; ++q) val[q] = (__bf16)0.f;
            }
            *(v8bf*)&bt[n][j] = val;
        }
        __syncthreads();

        // ----------------- 2 k-steps x MT M-tiles WMMAs -----------------
        const int mrow = lane & 15;
        const int koff = (lane & 16) ? 8 : 0;
        const int nn   = lane & 15;
        const int kbh  = (lane & 16) ? 8 : 0;
#pragma unroll
        for (int kstep = 0; kstep < 2; ++kstep) {
            const int kb = kstep * 32;
            v8bf blo = *(const v8bf*)&bt[nn][kb + kbh];
            v8bf bhi = *(const v8bf*)&bt[nn][kb + kbh + 16];
            v16bf bfrag = cat8(blo, bhi);
#pragma unroll
            for (int t = 0; t < MT; ++t) {
                const __bf16* ap = A + (size_t)(m0 + t * 16 + mrow) * K + kk + kb + koff;
                v16bf af = cat8(*(const v8bf*)ap, *(const v8bf*)(ap + 16));
                acc[t] = __builtin_amdgcn_wmma_f32_16x16x32_bf16(
                    false, af, false, bfrag, (short)0, acc[t], false, false);
            }
        }
    }

    // ------------------------------- epilogue -------------------------------
    const int col = n0 + (lane & 15);
#pragma unroll
    for (int t = 0; t < MT; ++t) {
        const int row_base = m0 + t * 16 + ((lane & 16) ? 8 : 0);
#pragma unroll
        for (int r = 0; r < 8; ++r) {
            int row = row_base + r;
            float v = acc[t][r];
            if (MODE == 0) {
                v += (row < 256) ? bias[row] : bias2[row - 256];
                outf[((size_t)(bb * CRM) + row) * PIX + col] = v;
            } else if (MODE == 1) {
                v += bias[row];
                outf[((size_t)z * OD + row) * PIX + col] = v;
                outh[(size_t)z * KFC + (size_t)row * PIX + col] = (__bf16)fmaxf(v, 0.f);
            } else if (MODE == 2) {
                v += bias[row];
                outh[(size_t)z * KFC + (size_t)row * PIX + col] = (__bf16)fmaxf(v, 0.f);
            } else {
                if (col < nvalid) atomicAdd(outf + (size_t)col * OD + row, v);
            }
        }
    }
}

// ---------------------------------------------------------------------------
// 4) bias + L2 normalize, one vector (256) per block
// ---------------------------------------------------------------------------
__global__ void normalize_kernel(const float* __restrict__ fcacc,
                                 const float* __restrict__ bias,
                                 float* __restrict__ out, int nvec) {
    __shared__ float red[256];
    int n = blockIdx.x;
    int o = threadIdx.x;
    if (n >= nvec) return;
    float v = fcacc[(size_t)n * OD + o] + bias[o];
    red[o] = v * v;
    __syncthreads();
    for (int s = 128; s > 0; s >>= 1) {
        if (o < s) red[o] += red[o + s];
        __syncthreads();
    }
    out[(size_t)n * OD + o] = v / sqrtf(red[0]);
}

// ---------------------------------------------------------------------------
extern "C" void kernel_launch(void* const* d_in, const int* in_sizes, int n_in,
                              void* d_out, int out_size, void* d_ws, size_t ws_size,
                              hipStream_t stream) {
    const float* inp    = (const float*)d_in[0];
    const float* masks  = (const float*)d_in[2];
    const float* w_ctx  = (const float*)d_in[3];
    const float* b_ctx  = (const float*)d_in[4];
    const float* w_rel  = (const float*)d_in[5];
    const float* b_rel  = (const float*)d_in[6];
    const float* w_objf = (const float*)d_in[7];
    const float* b_objf = (const float*)d_in[8];
    const float* w_relf = (const float*)d_in[9];
    const float* b_relf = (const float*)d_in[10];
    const float* w_fc   = (const float*)d_in[11];
    const float* b_fc   = (const float*)d_in[12];
    float* outp = (float*)d_out;

    char* ws = (char*)d_ws;
    size_t off = 0;
    float*  m_res   = (float*)(ws + off); off += 16896u * 4;            // B*S*384
    float*  tmp1    = (float*)(ws + off); off += 901120u * 4;           // B*S*320*64
    float*  tmp2    = (float*)(ws + off); off += 180224u * 4;           // B*S*64*64
    float*  tmp3    = (float*)(ws + off); off += 67584u * 4;            // B*S*64*24
    __bf16* Wcr     = (__bf16*)(ws + off); off += 163840u * 2;          // 640x256
    __bf16* Wobjf   = (__bf16*)(ws + off); off += 131072u * 2;          // 256x512
    __bf16* Wrelf   = (__bf16*)(ws + off); off += 229376u * 2;          // 256x896
    __bf16* Wfc     = (__bf16*)(ws + off); off += 25165824u * 2;        // 256x98304
    float*  ctxrel  = (float*)(ws + off); off += 983040u * 4;           // B*640*384
    float*  cobj    = (float*)(ws + off); off += 4325376u * 4;          // 44*256*384
    __bf16* oflat   = (__bf16*)(ws + off); off += 8650752u;             // 44*98304 bf16
    __bf16* rflat   = (__bf16*)(ws + off); off += 23789568u;            // 121*98304 bf16
    float*  fcacc   = (float*)(ws + off); off += 131072u;               // 128*256 f32

    // ---- mask resize: 480->64 (W), 320->64 (H), 64->24 (W), 64->16 (H) ----
    {
        int t;
        t = 901120; resample_kernel<<<(t + 255) / 256, 256, 0, stream>>>(masks, tmp1, BN * SN * 320, 480, 64, 1);
        t = 180224; resample_kernel<<<(t + 255) / 256, 256, 0, stream>>>(tmp1, tmp2, BN * SN, 320, 64, 64);
        t = 67584;  resample_kernel<<<(t + 255) / 256, 256, 0, stream>>>(tmp2, tmp3, BN * SN * 64, 64, 24, 1);
        t = 16896;  resample_kernel<<<(t + 255) / 256, 256, 0, stream>>>(tmp3, m_res, BN * SN, 64, 16, 24);
    }

    // ---- pack weights to bf16 ----
    pack_bf16_kernel<<<(65536 + 255) / 256, 256, 0, stream>>>(w_ctx, Wcr, 65536);
    pack_bf16_kernel<<<(98304 + 255) / 256, 256, 0, stream>>>(w_rel, Wcr + 65536, 98304);
    pack_bf16_kernel<<<(131072 + 255) / 256, 256, 0, stream>>>(w_objf, Wobjf, 131072);
    pack_bf16_kernel<<<(229376 + 255) / 256, 256, 0, stream>>>(w_relf, Wrelf, 229376);
    pack_bf16_kernel<<<(25165824 + 255) / 256, 256, 0, stream>>>(w_fc, Wfc, 25165824);

    dim3 blk(128);

    // ---- MODE 0: ctx+rel conv, all batches in grid.z (M=640 -> 5 WGs of 128 rows)
    gemm_wmma_kernel<0, 2><<<dim3(24, 5, BN), blk, 0, stream>>>(
        0, Wcr, 256, b_ctx, b_rel, inp, nullptr, nullptr, nullptr, nullptr,
        ctxrel, nullptr, 0, 0);

    // ---- MODE 1: obj conv for all (b,s) (M=256 -> one WG spans all 16 M-tiles)
    gemm_wmma_kernel<1, 4><<<dim3(24, 1, BN * SN), blk, 0, stream>>>(
        0, Wobjf, KOBJ, b_objf, nullptr, inp, ctxrel, nullptr, m_res, nullptr,
        cobj, oflat, 0, 0);

    // ---- FC for objects: split-K 64 chunks of 1536 ----
    hipMemsetAsync(fcacc, 0, 131072, stream);
    gemm_wmma_kernel<3, 4><<<dim3(3, 1, 64), blk, 0, stream>>>(
        0, Wfc, KFC, nullptr, nullptr, nullptr, nullptr, nullptr, nullptr, oflat,
        fcacc, nullptr, 1536, 44);
    normalize_kernel<<<44, 256, 0, stream>>>(fcacc, b_fc, outp, 44);

    // ---- per batch: rel conv (121 pairs) + FC + normalize ----
    for (int b = 0; b < BN; ++b) {
        gemm_wmma_kernel<2, 4><<<dim3(24, 1, SN * SN), blk, 0, stream>>>(
            b, Wrelf, KREL, b_relf, nullptr, nullptr, ctxrel, cobj, m_res, nullptr,
            nullptr, rflat, 0, 0);
        hipMemsetAsync(fcacc, 0, 131072, stream);
        gemm_wmma_kernel<3, 4><<<dim3(8, 1, 64), blk, 0, stream>>>(
            0, Wfc, KFC, nullptr, nullptr, nullptr, nullptr, nullptr, nullptr, rflat,
            fcacc, nullptr, 1536, 121);
        normalize_kernel<<<121, 256, 0, stream>>>(
            fcacc, b_fc, outp + 11264 + (size_t)b * 121 * 256, 121);
    }
}
